// GSLoRALinear_3925600109358
// MI455X (gfx1250) — compile-verified, matrix operand order checked by reading, not run on hardware
//
#include <hip/hip_runtime.h>

// ---------------------------------------------------------------------------
// GSLoRALinear for MI455X (gfx1250, wave32, WMMA)
//   out = x @ W^T + b + SUM_k alpha[b,k] * (x @ U[k]) @ V[k]^T
// bf16 V_WMMA_F32_16X16X32_BF16 with f32 accumulation.
//   kernel 1: repack U,V into bf16 [c,i] / [o,c] layouts (c = k*16+r)
//   kernel 2: y[m,c] = alpha[b, c>>4] * sum_i x[m,i] * U'[c,i]      (bf16 out)
//   kernel 3: out[m,o] = sum_i x[m,i]W[o,i] + sum_c y[m,c]V'[o,c] + b[o]
// Block tile 128(M) x 256(N), 8 waves, 64x64 per wave (4x4 WMMA accums).
// f32 operands: global_load_b128 -> v_cvt_pk_bf16_f32 -> LDS.
// bf16 operands: global_load_async_to_lds_b128 (ASYNCcnt) overlapped with WMMA.
// ---------------------------------------------------------------------------

typedef __attribute__((ext_vector_type(16))) __bf16 v16bf;
typedef __attribute__((ext_vector_type(8)))  __bf16 v8bf;
typedef __attribute__((ext_vector_type(8)))  float  v8f;

#define DIN   2048
#define DOUT  2048
#define SEQ   4096
#define NB    8
#define NC    256      // K*R = 16*16
#define MTOT  (NB * SEQ)
#define TS    40       // LDS row stride in bf16 elements (80 B; 16B-aligned chunks)
#define SCALING 1.0f   // lora_alpha / rank

// ---- gfx1250 async global->LDS (ASYNCcnt) ----------------------------------
__device__ __forceinline__ unsigned lds_addr32(const void* p) {
    // LDS aperture is 4GB-aligned: low 32 bits of the generic address are the
    // LDS byte address (ISA 10.2 aperture table).
    return (unsigned)(unsigned long long)p;
}
__device__ __forceinline__ void async_b128(unsigned lds, const void* gp) {
    asm volatile("global_load_async_to_lds_b128 %0, %1, off"
                 :: "v"(lds), "v"(gp) : "memory");
}
__device__ __forceinline__ void wait_async0() {
    asm volatile("s_wait_asynccnt 0x0" ::: "memory");
}

// ---- stage 128-row x 32-K f32 tile (A operand), convert -> bf16 ------------
__device__ __forceinline__ void stage_f32_A(const float* __restrict__ src, int ld,
                                            int row0, int k0, __bf16* dst, int tid) {
    const int row  = tid >> 1;           // 0..127
    const int half = (tid & 1) << 4;     // 0 or 16
    const float* p = src + (size_t)(row0 + row) * ld + (k0 + half);
    float4 f0 = ((const float4*)p)[0];
    float4 f1 = ((const float4*)p)[1];
    float4 f2 = ((const float4*)p)[2];
    float4 f3 = ((const float4*)p)[3];
    v8f a = {f0.x, f0.y, f0.z, f0.w, f1.x, f1.y, f1.z, f1.w};
    v8f b = {f2.x, f2.y, f2.z, f2.w, f3.x, f3.y, f3.z, f3.w};
    *(v8bf*)(dst + row * TS + half)     = __builtin_convertvector(a, v8bf);
    *(v8bf*)(dst + row * TS + half + 8) = __builtin_convertvector(b, v8bf);
}

// ---- stage 256-row x 32-K f32 tile (B operand), convert -> bf16 ------------
__device__ __forceinline__ void stage_f32_B(const float* __restrict__ src, int ld,
                                            int row0, int k0, __bf16* dst, int tid) {
    const int row = tid;                 // 0..255
    const float* p = src + (size_t)(row0 + row) * ld + k0;
#pragma unroll
    for (int h = 0; h < 2; ++h) {
        float4 f0 = ((const float4*)p)[4 * h + 0];
        float4 f1 = ((const float4*)p)[4 * h + 1];
        float4 f2 = ((const float4*)p)[4 * h + 2];
        float4 f3 = ((const float4*)p)[4 * h + 3];
        v8f a = {f0.x, f0.y, f0.z, f0.w, f1.x, f1.y, f1.z, f1.w};
        v8f b = {f2.x, f2.y, f2.z, f2.w, f3.x, f3.y, f3.z, f3.w};
        *(v8bf*)(dst + row * TS + 16 * h)     = __builtin_convertvector(a, v8bf);
        *(v8bf*)(dst + row * TS + 16 * h + 8) = __builtin_convertvector(b, v8bf);
    }
}

// ---- stage 128-row x 32-K bf16 tile (A operand) via async DMA --------------
__device__ __forceinline__ void stage_bf16_A_async(const __bf16* __restrict__ src, int ld,
                                                   int row0, int k0, __bf16* dst, int tid) {
    const int row  = tid >> 1;
    const int half = (tid & 1) << 4;
    const __bf16* p = src + (size_t)(row0 + row) * ld + (k0 + half);
    const unsigned l = lds_addr32(dst + row * TS + half);
    async_b128(l,       p);
    async_b128(l + 16u, p + 8);
}

// ---- stage 256-row x 32-K bf16 tile (B operand) via async DMA --------------
__device__ __forceinline__ void stage_bf16_B_async(const __bf16* __restrict__ src, int ld,
                                                   int row0, int k0, __bf16* dst, int tid) {
    const int row = tid;
    const __bf16* p = src + (size_t)(row0 + row) * ld + k0;
    const unsigned l = lds_addr32(dst + row * TS);
    async_b128(l,       p);
    async_b128(l + 16u, p + 8);
    async_b128(l + 32u, p + 16);
    async_b128(l + 48u, p + 24);
}

// ---- one 32-deep K-step for a wave's 64x64 sub-tile: 4x4 WMMAs -------------
// Per-lane fragment layout per CDNA5 ISA 7.12.2 (16-bit A 16x32):
//   lanes 0-15: row = lane, K = {0..7, 16..23}; lanes 16-31: K = {8..15, 24..31}
__device__ __forceinline__ void wave_mma(const __bf16* As, const __bf16* Bs,
                                         int lane, int wm, int wn, v8f acc[4][4]) {
    const int l15  = lane & 15;
    const int koff = (lane >> 4) << 3;   // 0 or 8
    v16bf a[4], b[4];
#pragma unroll
    for (int i = 0; i < 4; ++i) {
        const int row = wm * 64 + i * 16 + l15;
        v8bf lo = *(const v8bf*)(As + row * TS + koff);
        v8bf hi = *(const v8bf*)(As + row * TS + 16 + koff);
        a[i] = __builtin_shufflevector(lo, hi, 0,1,2,3,4,5,6,7,8,9,10,11,12,13,14,15);
    }
#pragma unroll
    for (int j = 0; j < 4; ++j) {
        const int col = wn * 64 + j * 16 + l15;
        v8bf lo = *(const v8bf*)(Bs + col * TS + koff);
        v8bf hi = *(const v8bf*)(Bs + col * TS + 16 + koff);
        b[j] = __builtin_shufflevector(lo, hi, 0,1,2,3,4,5,6,7,8,9,10,11,12,13,14,15);
    }
#pragma unroll
    for (int i = 0; i < 4; ++i)
#pragma unroll
        for (int j = 0; j < 4; ++j)
            acc[i][j] = __builtin_amdgcn_wmma_f32_16x16x32_bf16(
                false, a[i], false, b[j], (short)0, acc[i][j], false, false);
}

// ---------------------------------------------------------------------------
// Kernel 1: repack U (K,DIN,R) -> U'[c,i] bf16 and V (K,DOUT,R) -> V'[o,c] bf16
// ---------------------------------------------------------------------------
__global__ __launch_bounds__(256) void prep_kernel(const float* __restrict__ U,
                                                   const float* __restrict__ V,
                                                   unsigned short* __restrict__ Ubf_,
                                                   unsigned short* __restrict__ Vbf_) {
    __bf16* Ubf = (__bf16*)Ubf_;
    __bf16* Vbf = (__bf16*)Vbf_;
    const int id = blockIdx.x * 256 + threadIdx.x;      // 0 .. 512K-1
    {   // U'[c*DIN + i] = U[(k*DIN + i)*16 + r],  c = k*16+r
        const int c = id >> 11, i = id & 2047;
        const int k = c >> 4, r = c & 15;
        Ubf[id] = (__bf16)U[((size_t)k * DIN + i) * 16 + r];
    }
    {   // V'[o*256 + c] = V[(k*DOUT + o)*16 + r]
        const int o = id >> 8, c = id & 255;
        const int k = c >> 4, r = c & 15;
        Vbf[id] = (__bf16)V[((size_t)k * DOUT + o) * 16 + r];
    }
}

// ---------------------------------------------------------------------------
// Kernel 2: y[m,c] = alpha[b, c>>4] * sum_i x[m,i] * U'[c,i]  (M=32768,N=256,K=2048)
// ---------------------------------------------------------------------------
__global__ __launch_bounds__(256) void lora_xu_kernel(const float* __restrict__ x,
                                                      const float* __restrict__ alpha,
                                                      const unsigned short* __restrict__ Ubf_,
                                                      unsigned short* __restrict__ y_) {
    const __bf16* Ubf = (const __bf16*)Ubf_;
    __bf16* y = (__bf16*)y_;
    __shared__ __bf16 As[2][128 * TS];
    __shared__ __bf16 Bs[2][256 * TS];
    const int tid = threadIdx.x;
    const int m0 = blockIdx.x * 128;
    const int lane = tid & 31, w = tid >> 5, wm = w >> 2, wn = w & 3;

    v8f acc[4][4];
    const v8f vzero = {0.f, 0.f, 0.f, 0.f, 0.f, 0.f, 0.f, 0.f};
#pragma unroll
    for (int i = 0; i < 4; ++i)
#pragma unroll
        for (int j = 0; j < 4; ++j) acc[i][j] = vzero;

    stage_bf16_B_async(Ubf, DIN, 0, 0, Bs[0], tid);     // whole N=256 panel
    stage_f32_A(x, DIN, m0, 0, As[0], tid);
    wait_async0();
    __syncthreads();

#pragma unroll 1
    for (int kt = 0; kt < 64; ++kt) {
        const int cur = kt & 1, nxt = cur ^ 1;
        if (kt < 62)
            __builtin_prefetch(x + (size_t)(m0 + (tid >> 1)) * DIN + (kt + 2) * 32, 0, 1);
        if (kt < 63) {
            stage_bf16_B_async(Ubf, DIN, 0, (kt + 1) * 32, Bs[nxt], tid);
            stage_f32_A(x, DIN, m0, (kt + 1) * 32, As[nxt], tid);
        }
        wave_mma(As[cur], Bs[cur], lane, wm, wn, acc);  // DMA overlaps WMMA
        wait_async0();
        __syncthreads();
    }

    // epilogue: scale by alpha[b,k] (batch constant per tile since 128 | SEQ)
    const int b = m0 >> 12;            // m0 / SEQ
    const int l15 = lane & 15, lh = (lane >> 4) << 3;
#pragma unroll
    for (int j = 0; j < 4; ++j) {
        const int col = wn * 64 + j * 16 + l15;
        const float av = alpha[b * 16 + (col >> 4)] * SCALING;
#pragma unroll
        for (int i = 0; i < 4; ++i) {
            const int rowb = m0 + wm * 64 + i * 16 + lh;
#pragma unroll
            for (int v = 0; v < 8; ++v)
                y[(size_t)(rowb + v) * NC + col] = (__bf16)(acc[i][j][v] * av);
        }
    }
}

// ---------------------------------------------------------------------------
// Kernel 3: out[m,o] = sum_i x[m,i]W[o,i] + sum_c y[m,c]V'[o,c] + bias[o]
// Concatenated-K GEMM: 64 f32-converted tiles then 8 async-staged bf16 tiles.
// ---------------------------------------------------------------------------
__global__ __launch_bounds__(256) void gemm_main_kernel(const float* __restrict__ x,
                                                        const float* __restrict__ W,
                                                        const float* __restrict__ bias,
                                                        const unsigned short* __restrict__ y_,
                                                        const unsigned short* __restrict__ Vbf_,
                                                        float* __restrict__ out) {
    const __bf16* y   = (const __bf16*)y_;
    const __bf16* Vbf = (const __bf16*)Vbf_;
    __shared__ __bf16 As[2][128 * TS];
    __shared__ __bf16 Bs[2][256 * TS];
    const int tid = threadIdx.x;
    const int m0 = blockIdx.x * 128;
    const int n0 = blockIdx.y * 256;
    const int lane = tid & 31, w = tid >> 5, wm = w >> 2, wn = w & 3;

    v8f acc[4][4];
    const v8f vzero = {0.f, 0.f, 0.f, 0.f, 0.f, 0.f, 0.f, 0.f};
#pragma unroll
    for (int i = 0; i < 4; ++i)
#pragma unroll
        for (int j = 0; j < 4; ++j) acc[i][j] = vzero;

    stage_f32_A(x, DIN, m0, 0, As[0], tid);
    stage_f32_B(W, DIN, n0, 0, Bs[0], tid);
    __syncthreads();

    // phase 1: K = 0..2047 over x / W (f32 -> bf16 on the fly)
#pragma unroll 1
    for (int kt = 0; kt < 64; ++kt) {
        const int cur = kt & 1, nxt = cur ^ 1;
        if (kt < 62) {
            __builtin_prefetch(x + (size_t)(m0 + (tid >> 1)) * DIN + (kt + 2) * 32, 0, 1);
            __builtin_prefetch(W + (size_t)(n0 + tid) * DIN + (kt + 2) * 32, 0, 1);
        }
        if (kt < 63) {
            stage_f32_A(x, DIN, m0, (kt + 1) * 32, As[nxt], tid);
            stage_f32_B(W, DIN, n0, (kt + 1) * 32, Bs[nxt], tid);
        } else {        // stage first LoRA tile (already bf16) via async DMA
            stage_bf16_A_async(y, NC, m0, 0, As[nxt], tid);
            stage_bf16_B_async(Vbf, NC, n0, 0, Bs[nxt], tid);
        }
        wave_mma(As[cur], Bs[cur], lane, wm, wn, acc);
        wait_async0();
        __syncthreads();
    }

    // phase 2: concatenated K = 2048..2303 over y / V'  (8 tiles, starts buf 0)
#pragma unroll 1
    for (int kt = 0; kt < 8; ++kt) {
        const int cur = kt & 1, nxt = cur ^ 1;
        if (kt < 7) {
            stage_bf16_A_async(y, NC, m0, (kt + 1) * 32, As[nxt], tid);
            stage_bf16_B_async(Vbf, NC, n0, (kt + 1) * 32, Bs[nxt], tid);
        }
        wave_mma(As[cur], Bs[cur], lane, wm, wn, acc);
        wait_async0();
        __syncthreads();
    }

    // epilogue: + bias, f32 store (N-contiguous across lanes -> coalesced)
    const int l15 = lane & 15, lh = (lane >> 4) << 3;
#pragma unroll
    for (int j = 0; j < 4; ++j) {
        const int col = n0 + wn * 64 + j * 16 + l15;
        const float bv = bias[col];
#pragma unroll
        for (int i = 0; i < 4; ++i) {
            const int rowb = m0 + wm * 64 + i * 16 + lh;
#pragma unroll
            for (int v = 0; v < 8; ++v)
                out[(size_t)(rowb + v) * DOUT + col] = acc[i][j][v] + bv;
        }
    }
}

// ---------------------------------------------------------------------------
extern "C" void kernel_launch(void* const* d_in, const int* in_sizes, int n_in,
                              void* d_out, int out_size, void* d_ws, size_t ws_size,
                              hipStream_t stream) {
    const float* x     = (const float*)d_in[0];   // (8,4096,2048)
    const float* alpha = (const float*)d_in[1];   // (8,16)
    const float* W     = (const float*)d_in[2];   // (2048,2048)
    const float* bias  = (const float*)d_in[3];   // (2048,)
    const float* U     = (const float*)d_in[4];   // (16,2048,16)
    const float* V     = (const float*)d_in[5];   // (16,2048,16)
    float* out = (float*)d_out;

    // workspace (bf16 as u16): y[32768*256] | U'[256*2048] | V'[2048*256]
    unsigned short* yws = (unsigned short*)d_ws;
    unsigned short* Ubf = yws + (size_t)MTOT * NC;
    unsigned short* Vbf = Ubf + (size_t)NC * DIN;

    prep_kernel<<<dim3((NC * DIN) / 256), 256, 0, stream>>>(U, V, Ubf, Vbf);
    lora_xu_kernel<<<dim3(MTOT / 128), 256, 0, stream>>>(x, alpha, Ubf, yws);
    gemm_main_kernel<<<dim3(MTOT / 128, DOUT / 256), 256, 0, stream>>>(x, W, bias, yws, Vbf, out);
}
